// Residual_VQVAE_1640677507239
// MI455X (gfx1250) — matrix operand order repaired and checked by reading, not compiled, not run on hardware
//
#include <hip/hip_runtime.h>
#include <hip/hip_bf16.h>

typedef __attribute__((ext_vector_type(16))) _Float16 v16h;
typedef __attribute__((ext_vector_type(8)))  _Float16 v8h;
typedef __attribute__((ext_vector_type(8)))  float    v8f;
typedef __attribute__((ext_vector_type(4)))  unsigned v4u;
typedef __attribute__((ext_vector_type(8)))  int      v8i;
typedef __attribute__((ext_vector_type(4)))  int      v4i;
typedef __attribute__((address_space(3)))    _Float16 lds_h;

#define B_      256
#define T_      1632
#define T2_     816
#define T4_     408
#define HID_    64
#define CBD_    512
#define NTOK_   1024
#define NQ_     2
#define NT_     (B_ * T4_)       /* 104448 tokens */
#define NTILES_ (NT_ / 16)       /* 6528 */
#define NP_     (B_ * T2_)       /* 208896 positions (d2 tokens) */
#define NPTILES_ (NP_ / 16)      /* 13056 */

/* ---------- helpers ---------- */
__device__ inline v16h cat8(v8h a, v8h b) {
  return __builtin_shufflevector(a, b, 0,1,2,3,4,5,6,7,8,9,10,11,12,13,14,15);
}
/* async 16B copy: global -> LDS, tracked by ASYNCcnt */
__device__ inline void async_cp16(_Float16* lptr, const _Float16* gptr) {
  asm volatile("global_load_async_to_lds_b128 %0, %1, off"
               :: "v"((lds_h*)lptr), "v"(gptr)
               : "memory");
}
/* TDM 2D tile load global->LDS (f16 elems), LDS row padding via D# pad fields.
   td0 = elems/row, td1 = rows, stride0 = global row stride (elems).
   padi: pad interval code (7 => every 256 DWORDs = 1024B row)
   pada: pad amount code  (3 => 4 DWORDs = 8 halfs; 7 => 8 DWORDs = 16 halfs) */
__device__ __forceinline__ void tdm_load_2d(void* lds_dst, const void* gsrc,
    unsigned td0, unsigned td1, unsigned stride0,
    unsigned padi, unsigned pada, unsigned paden) {
#if __has_builtin(__builtin_amdgcn_tensor_load_to_lds)
  unsigned long long ga = (unsigned long long)gsrc;
  unsigned lds = (unsigned)(unsigned long long)
                 (__attribute__((address_space(3))) void*)lds_dst;
  v4u g0;
  g0[0] = 1u;                                        /* count=1, user mode   */
  g0[1] = lds;                                       /* lds_addr             */
  g0[2] = (unsigned)ga;                              /* global_addr[31:0]    */
  g0[3] = (unsigned)((ga >> 32) & 0x01FFFFFFu) | (2u << 30);  /* [56:32]+type */
  v8i g1;
  g1[0] = (int)((1u << 16) | (paden << 20) | (padi << 22) | (pada << 25));
  g1[1] = (int)((td0 & 0xFFFFu) << 16);              /* tensor_dim0 lo16     */
  g1[2] = (int)(((td0 >> 16) & 0xFFFFu) | ((td1 & 0xFFFFu) << 16)); /* d0hi|d1lo */
  g1[3] = (int)(((td1 >> 16) & 0xFFFFu) | ((td0 & 0xFFFFu) << 16)); /* d1hi|tile0 */
  g1[4] = (int)(td1 & 0xFFFFu);                      /* tile_dim1; tile_dim2=0 */
  g1[5] = (int)stride0;                              /* tensor_dim0_stride lo */
  g1[6] = 0;
  g1[7] = 0;
  v4i z4 = {0, 0, 0, 0};
  v8i z8 = {0, 0, 0, 0, 0, 0, 0, 0};
  /* clang-23 / amdgpu-toolchain 6-arg form */
  __builtin_amdgcn_tensor_load_to_lds(g0, g1, z4, z4, z8, 0);
#endif
}

/* ---------- tiny prep kernels ---------- */
__global__ void k_zero_acc(float* acc) {
  int i = threadIdx.x;
  if (i < 16) acc[i] = 0.f;
}
__global__ void k_f32_to_f16(const float* __restrict__ s, _Float16* __restrict__ d, long n) {
  long i = (long)blockIdx.x * blockDim.x + threadIdx.x;
  if (i < n) d[i] = (_Float16)s[i];
}
/* Wt_d1 [i=512][o=64][k=2] -> wd1t f16 [n=o*2+k][i=512] */
__global__ void k_wd1t(const float* __restrict__ w, _Float16* __restrict__ d) {
  int id = blockIdx.x * blockDim.x + threadIdx.x;   /* 65536 */
  if (id >= 512 * 128) return;
  int n = id & 127; int i = id >> 7;
  d[(long)n * 512 + i] = (_Float16)w[(long)i * 128 + n];
}
/* W_e2 [o=64][i=64][k=2] -> we2t f16 [o][j=k*64+i] */
__global__ void k_we2t(const float* __restrict__ w, _Float16* __restrict__ d) {
  int id = blockIdx.x * blockDim.x + threadIdx.x;   /* 8192 */
  if (id >= 64 * 128) return;
  int o = id >> 7; int j = id & 127; int k = j >> 6; int i = j & 63;
  d[o * 128 + j] = (_Float16)w[o * 128 + i * 2 + k];
}
/* Wt_d2 [i=64][o=64][k=2] -> wd2t f16 [n=o*2+k][i=64] */
__global__ void k_wd2t(const float* __restrict__ w, _Float16* __restrict__ d) {
  int id = blockIdx.x * blockDim.x + threadIdx.x;   /* 8192 */
  if (id >= 128 * 64) return;
  int n = id >> 6; int i = id & 63;
  d[n * 64 + i] = (_Float16)w[i * 128 + n];
}
__global__ void k_cb_norms(const float* __restrict__ cb, float* __restrict__ cbn) {
  long n = blockIdx.x;                               /* 2048 rows */
  int tid = threadIdx.x;
  const float* r = cb + n * 512;
  float p = 0.f;
  for (int d = tid; d < 512; d += 256) { float v = r[d]; p += v * v; }
  __shared__ float red[256];
  red[tid] = p; __syncthreads();
  for (int st = 128; st; st >>= 1) { if (tid < st) red[tid] += red[tid + st]; __syncthreads(); }
  if (tid == 0) cbn[n] = red[0];
}

/* ---------- encoder ---------- */
__global__ void k_enc1(const float* __restrict__ img, const float* __restrict__ w,
                       const float* __restrict__ bias, _Float16* __restrict__ h1h) {
  long id = (long)blockIdx.x * blockDim.x + threadIdx.x;  /* B*T2*64 */
  if (id >= (long)B_ * T2_ * HID_) return;
  int o = (int)(id & 63); long bt = id >> 6;
  int t = (int)(bt % T2_); long b = bt / T2_;
  float x0 = img[b * T_ + 2 * t], x1 = img[b * T_ + 2 * t + 1];
  float v = x0 * w[o * 2] + x1 * w[o * 2 + 1] + bias[o];
  h1h[(b * T2_ + t) * 64 + o] = (_Float16)(v > 0.f ? v : 0.f);
}

/* conv2 k2s2 via WMMA: A = 16 tokens x 128 (h1 row pair), B = we2t 128x64 */
__global__ __launch_bounds__(32) void k_enc2_wmma(const _Float16* __restrict__ h1h,
    const _Float16* __restrict__ we2t, const float* __restrict__ bias,
    _Float16* __restrict__ h2h) {
  __shared__ _Float16 sA[16 * 136];
  __shared__ _Float16 sB[64 * 144];
  int lane = threadIdx.x;
  long tok0 = (long)blockIdx.x * 16;
  #pragma unroll
  for (int i = 0; i < 8; ++i) {
    int ch = lane + 32 * i;              /* 16 rows x 16 chunks */
    int r = ch >> 4, c = (ch & 15) * 8;
    long tok = tok0 + r; long b = tok / T4_; int t = (int)(tok % T4_);
    async_cp16(&sA[r * 136 + c], h1h + (b * T2_ + 2 * t) * 64 + c);
  }
  #pragma unroll 4
  for (int i = 0; i < 32; ++i) {
    int ch = lane + 32 * i;              /* 64 rows x 16 chunks */
    int r = ch >> 4, c = (ch & 15) * 8;
    async_cp16(&sB[r * 144 + c], we2t + (long)r * 128 + c);
  }
  asm volatile("s_wait_asynccnt 0x0" ::: "memory");
  __syncthreads();
  int half = lane >> 4, ln = lane & 15;
  for (int nt = 0; nt < 4; ++nt) {
    v8f acc = {0.f,0.f,0.f,0.f,0.f,0.f,0.f,0.f};
    #pragma unroll
    for (int kb = 0; kb < 128; kb += 32) {
      v16h a = cat8(*(const v8h*)&sA[ln * 136 + kb + half * 8],
                    *(const v8h*)&sA[ln * 136 + kb + 16 + half * 8]);
      v16h bm = cat8(*(const v8h*)&sB[(nt * 16 + ln) * 144 + kb + half * 8],
                     *(const v8h*)&sB[(nt * 16 + ln) * 144 + kb + 16 + half * 8]);
      acc = __builtin_amdgcn_wmma_f32_16x16x32_f16(false, a, false, bm,
                                                   (short)0, acc, false, false);
    }
    int o = nt * 16 + ln;
    float bi = bias[o];
    #pragma unroll
    for (int j = 0; j < 8; ++j) {
      long tok = tok0 + j + half * 8;
      float v = acc[j] + bi; v = v > 0.f ? v : 0.f;
      h2h[tok * 64 + o] = (_Float16)v;
    }
  }
}

/* conv1x1 e3 via WMMA; A async from h2h; writes f32 xres + f16 xh, zeroes quant */
__global__ __launch_bounds__(32) void k_e3_wmma(const _Float16* __restrict__ h2h,
    const _Float16* __restrict__ w3h, const float* __restrict__ b3,
    float* __restrict__ xres, float* __restrict__ quant, _Float16* __restrict__ xh) {
  __shared__ _Float16 sa[16 * 72];
  int lane = threadIdx.x;
  long tok0 = (long)blockIdx.x * 16;
  #pragma unroll
  for (int i = 0; i < 4; ++i) {
    int ch = lane + 32 * i;              /* 16 rows x 8 chunks */
    int r = ch >> 3, c = (ch & 7) * 8;
    async_cp16(&sa[r * 72 + c], h2h + (tok0 + r) * 64 + c);
  }
  asm volatile("s_wait_asynccnt 0x0" ::: "memory");
  __syncthreads();
  int half = lane >> 4, ln = lane & 15;
  for (int nt = 0; nt < 32; ++nt) {
    v8f acc = {0.f,0.f,0.f,0.f,0.f,0.f,0.f,0.f};
    for (int kb = 0; kb < 64; kb += 32) {
      v16h a = cat8(*(const v8h*)&sa[ln * 72 + kb + half * 8],
                    *(const v8h*)&sa[ln * 72 + kb + 16 + half * 8]);
      v16h bm = cat8(*(const v8h*)&w3h[(long)(nt * 16 + ln) * 64 + kb + half * 8],
                     *(const v8h*)&w3h[(long)(nt * 16 + ln) * 64 + kb + 16 + half * 8]);
      acc = __builtin_amdgcn_wmma_f32_16x16x32_f16(false, a, false, bm,
                                                   (short)0, acc, false, false);
    }
    int c = nt * 16 + ln;
    float bias = b3[c];
    #pragma unroll
    for (int j = 0; j < 8; ++j) {
      long tok = tok0 + j + half * 8;
      float v = acc[j] + bias;
      xres[tok * 512 + c]  = v;
      xh[tok * 512 + c]    = (_Float16)v;
      quant[tok * 512 + c] = 0.f;
    }
  }
}

/* ---------- VQ distance + argmin: 4 waves/block, TDM double-buffered codebook ---------- */
__global__ __launch_bounds__(128) void k_vq_dist4(const _Float16* __restrict__ xh,
    const _Float16* __restrict__ cbh, const float* __restrict__ cbn,
    int* __restrict__ idxo) {
  __shared__ _Float16 sA[4 * 16 * 520];
  __shared__ _Float16 sB[2 * 16 * 528];
  int tid = threadIdx.x;
  int wv = tid >> 5, lane = tid & 31;
  long tok0 = ((long)blockIdx.x * 4 + wv) * 16;
  _Float16* Aw = &sA[wv * 16 * 520];

  /* per-wave A tile via ASYNCcnt path */
  #pragma unroll 4
  for (int i = 0; i < 32; ++i) {
    int ch = lane + 32 * i;                   /* 16 rows x 64 chunks */
    int r = ch >> 6, c = (ch & 63) * 8;
    async_cp16(&Aw[r * 520 + c], xh + (tok0 + r) * 512 + c);
  }
  asm volatile("s_wait_asynccnt 0x0" ::: "memory");

  /* shared B tile: wave 0 issues one TDM load per tile (pad 512->528 halfs) */
  auto fillB = [&](int nt, int buf) {
    tdm_load_2d(&sB[buf * 16 * 528], cbh + (long)nt * 16 * 512,
                512, 16, 512, 7, 7, 1);
  };
  if (wv == 0) fillB(0, 0);

  int half = lane >> 4, ln = lane & 15;
  float best[8]; int bid[8];
  #pragma unroll
  for (int j = 0; j < 8; ++j) { best[j] = 3.4e38f; bid[j] = 0; }

  for (int nt = 0; nt < 64; ++nt) {
    int buf = nt & 1;
    if (nt + 1 < 64) {
      __syncthreads();                        /* buf^1 free for overwrite      */
      if (wv == 0) {
        fillB(nt + 1, buf ^ 1);
        asm volatile("s_wait_tensorcnt 0x1" ::: "memory");  /* cur buf ready */
      }
    } else {
      if (wv == 0) asm volatile("s_wait_tensorcnt 0x0" ::: "memory");
    }
    __syncthreads();                          /* fill visible to all waves     */
    const _Float16* Bt = &sB[buf * 16 * 528];
    v8f acc = {0.f,0.f,0.f,0.f,0.f,0.f,0.f,0.f};
    #pragma unroll 4
    for (int kb = 0; kb < 512; kb += 32) {
      v16h a = cat8(*(const v8h*)&Aw[ln * 520 + kb + half * 8],
                    *(const v8h*)&Aw[ln * 520 + kb + 16 + half * 8]);
      v16h bm = cat8(*(const v8h*)&Bt[ln * 528 + kb + half * 8],
                     *(const v8h*)&Bt[ln * 528 + kb + 16 + half * 8]);
      acc = __builtin_amdgcn_wmma_f32_16x16x32_f16(false, a, false, bm,
                                                   (short)0, acc, false, false);
    }
    int n = nt * 16 + ln;
    float cn = cbn[n];
    #pragma unroll
    for (int j = 0; j < 8; ++j) {
      float d = cn - 2.f * acc[j];
      if (d < best[j]) { best[j] = d; bid[j] = n; }
    }
  }
  #pragma unroll
  for (int j = 0; j < 8; ++j) {
    #pragma unroll
    for (int off = 8; off; off >>= 1) {
      float ob = __shfl_xor(best[j], off, 16);
      int   oi = __shfl_xor(bid[j],  off, 16);
      if (ob < best[j] || (ob == best[j] && oi < bid[j])) { best[j] = ob; bid[j] = oi; }
    }
  }
  if (ln == 0) {
    #pragma unroll
    for (int j = 0; j < 8; ++j) idxo[tok0 + j + half * 8] = bid[j];
  }
}

/* gather + residual update + commit-loss partial; maintains f16 mirrors */
__global__ void k_vq_update(const float* __restrict__ cb, const int* __restrict__ idxw,
    float* __restrict__ xres, float* __restrict__ quant, float* __restrict__ acc,
    int q, int* __restrict__ oidx, _Float16* __restrict__ xh, _Float16* __restrict__ qh) {
  long tok = blockIdx.x;
  int tid = threadIdx.x;
  int id = idxw[tok];
  const float* crow = cb + ((long)q * NTOK_ + id) * 512;
  float part = 0.f;
  for (int d = tid; d < 512; d += 256) {
    long td = tok * 512 + d;
    float qv = crow[d];
    float rv = xres[td];
    float qn = quant[td] + qv;
    quant[td] = qn;
    qh[td] = (_Float16)qn;
    float nr = rv - qv;
    xres[td] = nr;
    xh[td] = (_Float16)nr;
    float df = qv - rv; part += df * df;
  }
  __shared__ float red[256];
  red[tid] = part; __syncthreads();
  for (int st = 128; st; st >>= 1) { if (tid < st) red[tid] += red[tid + st]; __syncthreads(); }
  if (tid == 0) { atomicAdd(acc + 1 + q, red[0]); oidx[tok * 2 + q] = id; }
}

/* ---------- decoder ---------- */
/* convT d1 via WMMA; A (pad->520) and B (pad->528) staged by TDM */
__global__ __launch_bounds__(32) void k_d1_wmma(const _Float16* __restrict__ qh,
    const _Float16* __restrict__ wd1t, const float* __restrict__ b1,
    _Float16* __restrict__ d1h) {
  __shared__ _Float16 sA[16 * 520];
  __shared__ _Float16 sB[2 * 16 * 528];
  int lane = threadIdx.x;
  long tok0 = (long)blockIdx.x * 16;
  tdm_load_2d(sA, qh + tok0 * 512, 512, 16, 512, 7, 3, 1);
  tdm_load_2d(&sB[0], wd1t, 512, 16, 512, 7, 7, 1);
  int half = lane >> 4, ln = lane & 15;
  for (int nt = 0; nt < 8; ++nt) {
    int buf = nt & 1;
    if (nt + 1 < 8) {
      tdm_load_2d(&sB[(buf ^ 1) * 16 * 528], wd1t + (long)(nt + 1) * 16 * 512,
                  512, 16, 512, 7, 7, 1);
      asm volatile("s_wait_tensorcnt 0x1" ::: "memory");
    } else {
      asm volatile("s_wait_tensorcnt 0x0" ::: "memory");
    }
    const _Float16* Bt = &sB[buf * 16 * 528];
    v8f acc = {0.f,0.f,0.f,0.f,0.f,0.f,0.f,0.f};
    #pragma unroll 4
    for (int kb = 0; kb < 512; kb += 32) {
      v16h a = cat8(*(const v8h*)&sA[ln * 520 + kb + half * 8],
                    *(const v8h*)&sA[ln * 520 + kb + 16 + half * 8]);
      v16h bm = cat8(*(const v8h*)&Bt[ln * 528 + kb + half * 8],
                     *(const v8h*)&Bt[ln * 528 + kb + 16 + half * 8]);
      acc = __builtin_amdgcn_wmma_f32_16x16x32_f16(false, a, false, bm,
                                                   (short)0, acc, false, false);
    }
    int n = nt * 16 + ln; int o = n >> 1, k = n & 1;
    float bias = b1[o];
    #pragma unroll
    for (int j = 0; j < 8; ++j) {
      long tok = tok0 + j + half * 8;
      long b = tok / T4_; int tt = (int)(tok % T4_);
      float v = acc[j] + bias; v = v > 0.f ? v : 0.f;
      d1h[(b * T2_ + 2 * tt + k) * 64 + o] = (_Float16)v;
    }
  }
}

/* convT d2 via WMMA: A = 16 positions x 64 (d1h), B = wd2t 64x128 */
__global__ __launch_bounds__(32) void k_d2_wmma(const _Float16* __restrict__ d1h,
    const _Float16* __restrict__ wd2t, const float* __restrict__ bias,
    float* __restrict__ d2b) {
  __shared__ _Float16 sA[16 * 72];
  __shared__ _Float16 sB[128 * 64];
  int lane = threadIdx.x;
  long tok0 = (long)blockIdx.x * 16;   /* position tokens (b*T2+p) */
  #pragma unroll
  for (int i = 0; i < 4; ++i) {
    int ch = lane + 32 * i; int r = ch >> 3, c = (ch & 7) * 8;
    async_cp16(&sA[r * 72 + c], d1h + (tok0 + r) * 64 + c);
  }
  #pragma unroll 4
  for (int i = 0; i < 32; ++i) {
    int ch = lane + 32 * i;              /* 128 rows x 8 chunks */
    int r = ch >> 3, c = (ch & 7) * 8;
    async_cp16(&sB[r * 64 + c], wd2t + (long)r * 64 + c);
  }
  asm volatile("s_wait_asynccnt 0x0" ::: "memory");
  __syncthreads();
  int half = lane >> 4, ln = lane & 15;
  for (int nt = 0; nt < 8; ++nt) {
    v8f acc = {0.f,0.f,0.f,0.f,0.f,0.f,0.f,0.f};
    #pragma unroll
    for (int kb = 0; kb < 64; kb += 32) {
      v16h a = cat8(*(const v8h*)&sA[ln * 72 + kb + half * 8],
                    *(const v8h*)&sA[ln * 72 + kb + 16 + half * 8]);
      v16h bm = cat8(*(const v8h*)&sB[(nt * 16 + ln) * 64 + kb + half * 8],
                     *(const v8h*)&sB[(nt * 16 + ln) * 64 + kb + 16 + half * 8]);
      acc = __builtin_amdgcn_wmma_f32_16x16x32_f16(false, a, false, bm,
                                                   (short)0, acc, false, false);
    }
    int n = nt * 16 + ln; int o = n >> 1, k = n & 1;
    float bi = bias[o];
    #pragma unroll
    for (int j = 0; j < 8; ++j) {
      long tok = tok0 + j + half * 8;
      long b = tok / T2_; int p = (int)(tok % T2_);
      float v = acc[j] + bi; v = v > 0.f ? v : 0.f;
      d2b[(b * T_ + 2 * p + k) * 64 + o] = v;
    }
  }
}

/* conv1x1 d3 + recon loss partial + write out */
__global__ void k_d3_loss(const float* __restrict__ d2, const float* __restrict__ w,
                          const float* __restrict__ b3, const float* __restrict__ img,
                          float* __restrict__ out, float* __restrict__ acc) {
  long id = (long)blockIdx.x * blockDim.x + threadIdx.x;   /* B*T = 417792 */
  int tid = threadIdx.x;
  const float* xr = d2 + id * 64;
  float s = b3[0];
  for (int i = 0; i < 64; ++i) s += xr[i] * w[i];
  out[id] = s;
  float df = img[id] - s; df *= df;
  __shared__ float red[256];
  red[tid] = df; __syncthreads();
  for (int st = 128; st; st >>= 1) { if (tid < st) red[tid] += red[tid + st]; __syncthreads(); }
  if (tid == 0) atomicAdd(acc, red[0]);
}
__global__ void k_write_quant(const float* __restrict__ quant, float* __restrict__ oq) {
  long id = (long)blockIdx.x * blockDim.x + threadIdx.x;   /* NT*512 */
  if (id >= (long)NT_ * 512) return;
  long tok = id >> 9; int c = (int)(id & 511);
  long b = tok / T4_; int tt = (int)(tok % T4_);
  oq[(b * 512 + c) * T4_ + tt] = quant[id];
}
__global__ void k_finalize(const float* __restrict__ acc, float* __restrict__ out) {
  out[(long)B_ * T_]     = acc[0] / (float)((long)B_ * T_);
  out[(long)B_ * T_ + 1] = acc[1] / (float)((long)NT_ * 512);
  out[(long)B_ * T_ + 2] = acc[2] / (float)((long)NT_ * 512);
}

/* ---------- launcher ---------- */
extern "C" void kernel_launch(void* const* d_in, const int* in_sizes, int n_in,
                              void* d_out, int out_size, void* d_ws, size_t ws_size,
                              hipStream_t stream) {
  const float* img   = (const float*)d_in[0];
  const float* W_e1  = (const float*)d_in[1];
  const float* b_e1  = (const float*)d_in[2];
  const float* W_e2  = (const float*)d_in[3];
  const float* b_e2  = (const float*)d_in[4];
  const float* W_e3  = (const float*)d_in[5];
  const float* b_e3  = (const float*)d_in[6];
  const float* cb    = (const float*)d_in[7];
  const float* Wt_d1 = (const float*)d_in[8];
  const float* b_d1  = (const float*)d_in[9];
  const float* Wt_d2 = (const float*)d_in[10];
  const float* b_d2  = (const float*)d_in[11];
  const float* W_d3  = (const float*)d_in[12];
  const float* b_d3  = (const float*)d_in[13];

  char* w = (char*)d_ws;
  float* xres  = (float*)w; w += (size_t)NT_ * 512 * 4;
  float* quant = (float*)w; w += (size_t)NT_ * 512 * 4;
  float* d2b   = (float*)w; w += (size_t)B_ * T_ * 64 * 4;
  _Float16* h1h = (_Float16*)w; w += (size_t)B_ * T2_ * 64 * 2;
  _Float16* h2h = (_Float16*)w; w += (size_t)B_ * T4_ * 64 * 2;
  _Float16* d1h = (_Float16*)w; w += (size_t)B_ * T2_ * 64 * 2;
  _Float16* xh  = (_Float16*)w; w += (size_t)NT_ * 512 * 2;
  _Float16* qh  = (_Float16*)w; w += (size_t)NT_ * 512 * 2;
  _Float16* cbh = (_Float16*)w; w += (size_t)NQ_ * NTOK_ * 512 * 2;
  float* cbn    = (float*)w;    w += (size_t)NQ_ * NTOK_ * 4;
  _Float16* w3h  = (_Float16*)w; w += (size_t)512 * 64 * 2;
  _Float16* wd1t = (_Float16*)w; w += (size_t)128 * 512 * 2;
  _Float16* we2t = (_Float16*)w; w += (size_t)64 * 128 * 2;
  _Float16* wd2t = (_Float16*)w; w += (size_t)128 * 64 * 2;
  int* idxw      = (int*)w;      w += (size_t)NQ_ * NT_ * 4;
  float* acc     = (float*)w;    w += 64;

  float* out  = (float*)d_out;
  int*   oidx = (int*)(out + (long)B_ * T_ + 3);
  float* oq   = out + (long)B_ * T_ + 3 + (long)NT_ * NQ_;

  hipLaunchKernelGGL(k_zero_acc, dim3(1), dim3(64), 0, stream, acc);
  hipLaunchKernelGGL(k_f32_to_f16, dim3((512 * 64 + 255) / 256), dim3(256), 0, stream,
                     W_e3, w3h, (long)512 * 64);
  hipLaunchKernelGGL(k_f32_to_f16, dim3((NQ_ * NTOK_ * 512 + 255) / 256), dim3(256), 0, stream,
                     cb, cbh, (long)NQ_ * NTOK_ * 512);
  hipLaunchKernelGGL(k_wd1t, dim3(256), dim3(256), 0, stream, Wt_d1, wd1t);
  hipLaunchKernelGGL(k_we2t, dim3(32), dim3(256), 0, stream, W_e2, we2t);
  hipLaunchKernelGGL(k_wd2t, dim3(32), dim3(256), 0, stream, Wt_d2, wd2t);
  hipLaunchKernelGGL(k_cb_norms, dim3(NQ_ * NTOK_), dim3(256), 0, stream, cb, cbn);

  hipLaunchKernelGGL(k_enc1, dim3((B_ * T2_ * 64) / 256), dim3(256), 0, stream,
                     img, W_e1, b_e1, h1h);
  hipLaunchKernelGGL(k_enc2_wmma, dim3(NTILES_), dim3(32), 0, stream,
                     h1h, we2t, b_e2, h2h);
  hipLaunchKernelGGL(k_e3_wmma, dim3(NTILES_), dim3(32), 0, stream,
                     h2h, w3h, b_e3, xres, quant, xh);

  for (int q = 0; q < NQ_; ++q) {
    hipLaunchKernelGGL(k_vq_dist4, dim3(NTILES_ / 4), dim3(128), 0, stream,
                       xh, cbh + (size_t)q * NTOK_ * 512, cbn + (size_t)q * NTOK_,
                       idxw + (size_t)q * NT_);
    hipLaunchKernelGGL(k_vq_update, dim3(NT_), dim3(256), 0, stream,
                       cb, idxw + (size_t)q * NT_, xres, quant, acc, q, oidx, xh, qh);
  }

  hipLaunchKernelGGL(k_d1_wmma, dim3(NTILES_), dim3(32), 0, stream, qh, wd1t, b_d1, d1h);
  hipLaunchKernelGGL(k_d2_wmma, dim3(NPTILES_), dim3(32), 0, stream,
                     d1h, wd2t, b_d2, d2b);
  hipLaunchKernelGGL(k_d3_loss, dim3((B_ * T_) / 256), dim3(256), 0, stream,
                     d2b, W_d3, b_d3, img, out, acc);
  hipLaunchKernelGGL(k_write_quant, dim3(((long)NT_ * 512) / 256), dim3(256), 0, stream,
                     quant, oq);
  hipLaunchKernelGGL(k_finalize, dim3(1), dim3(1), 0, stream, acc, out);
}
